// CrossFormerBlock_69655779606846
// MI455X (gfx1250) — compile-verified, matrix operand order checked by reading, not run on hardware
//
#include <hip/hip_runtime.h>
#include <hip/hip_bf16.h>
#include <math.h>

// ---------------------------------------------------------------------------
// CrossFormerBlock for MI455X (gfx1250, wave32, WMMA 16x16x32 f16 -> f32 acc)
// Data movement: GLOBAL_LOAD_ASYNC_TO_LDS_B128 for all LDS staging (ASYNCcnt),
// WMMA for all five GEMMs, VALU for softmax/LN/conv.
// ---------------------------------------------------------------------------

typedef __attribute__((ext_vector_type(16))) _Float16 v16h;
typedef __attribute__((ext_vector_type(8)))  float    v8f;

// K-index inside a 16x32 f16 A/B fragment for element e (0..15) of the v16h,
// given lane-half hi = lane/16 (CDNA5 ISA 7.12.2 16-bit A/B layout).
__device__ __forceinline__ int kmap(int e, int hi) {
    return ((e >> 3) << 4) + (((e >> 1) & 3) << 1) + (hi << 3) + (e & 1);
}

// Wave-local LDS producer->consumer sync (same-wave lane exchange via LDS).
__device__ __forceinline__ void wave_lds_sync() {
    __builtin_amdgcn_wave_barrier();
    asm volatile("s_wait_dscnt 0x0" ::: "memory");
    __builtin_amdgcn_wave_barrier();
}

// Async 16B global->LDS copy (no VGPR data path; tracked by ASYNCcnt).
// LDS address = low 32 bits of the flat shared pointer (ISA LDS aperture rule).
__device__ __forceinline__ void async_copy_b128(void* lds_dst, const void* gsrc) {
    unsigned l = (unsigned)(unsigned long long)lds_dst;
    asm volatile("global_load_async_to_lds_b128 %0, %1, off"
                 :: "v"(l), "v"(gsrc) : "memory");
}
__device__ __forceinline__ void wait_async() {
    asm volatile("s_wait_asynccnt 0x0" ::: "memory");
}

// Static problem dims
#define CC   128
#define TT   4
#define GG   4
#define NTOK 256        // tokens per window (G^3 * T)
#define NWIN 256        // windows
#define NHD  8          // heads
#define HDSZ 16         // head dim
#define MLPH 512
#define NROW 65536      // total tokens B*D*H*W*T
#define ATT_SCALE 0.25f // 1/sqrt(16)

// ---------------------------------------------------------------------------
// Kernel 1: LN + pos/time embed -> LDS f16, then QKV GEMM (256x384, K=128)
// ---------------------------------------------------------------------------
__global__ __launch_bounds__(256) void cf_k1_ln_pe_qkv(
    const float* __restrict__ x, const float* __restrict__ n1g,
    const float* __restrict__ n1b, const float* __restrict__ pos,
    const float* __restrict__ temb, const float* __restrict__ qkv_w,
    const float* __restrict__ qkv_b,
    _Float16* __restrict__ Q, _Float16* __restrict__ K, _Float16* __restrict__ V)
{
    __shared__ _Float16 sh[NTOK * CC];   // 64 KiB (LDS is 320 KiB/WGP)
    const int tid = threadIdx.x, win = blockIdx.x;

    // ---- per-token LN + embeds (1 thread per token) ----
    {
        const int n = tid;
        const int t = n >> 6, rem = n & 63;
        const int gd = rem >> 4, gh = (rem >> 2) & 3, gw = rem & 3;
        const int wd = win >> 6, wh = (win >> 3) & 7, ww = win & 7;
        const int d = wd * GG + gd, h = wh * GG + gh, w = ww * GG + gw;
        const float* xr = x + ((((size_t)d * 32 + h) * 32 + w) * TT + t) * CC;
        float m = 0.f, s2 = 0.f;
        for (int c = 0; c < CC; ++c) { float v = xr[c]; m += v; s2 += v * v; }
        m *= (1.f / CC);
        const float inv = rsqrtf(s2 * (1.f / CC) - m * m + 1e-5f);
        const float* pr = pos + ((gd * GG + gh) * GG + gw) * CC;
        const float* tr = temb + t * CC;
        _Float16* dst = sh + n * CC;
        for (int c = 0; c < CC; ++c) {
            float hv = (xr[c] - m) * inv * n1g[c] + n1b[c] + pr[c] + tr[c];
            dst[c] = (_Float16)hv;
        }
    }
    __syncthreads();

    // ---- GEMM: sh(256x128) @ qkv_w^T(128x384) ----
    const int wave = tid >> 5, lane = tid & 31, lr = lane & 15, hi = lane >> 4;
    for (int idx = wave; idx < 16 * 24; idx += 8) {
        const int rt = idx / 24, ct = idx % 24;
        v8f acc = {};
        for (int kk = 0; kk < 4; ++kk) {
            v16h a, bf;
            for (int e = 0; e < 16; ++e) {
                const int k = kmap(e, hi);
                a[e]  = sh[(rt * 16 + lr) * CC + kk * 32 + k];
                bf[e] = (_Float16)qkv_w[(size_t)(ct * 16 + lr) * CC + kk * 32 + k];
            }
            acc = __builtin_amdgcn_wmma_f32_16x16x32_f16(
                false, a, false, bf, (short)0, acc, false, false);
        }
        for (int r = 0; r < 8; ++r) {
            const int nrow = rt * 16 + r + 8 * hi;
            const int o = ct * 16 + lr;
            float val = acc[r] + qkv_b[o];
            const int oc = o & 127;
            const int head = oc >> 4, hd = oc & 15;
            const size_t addr = (((size_t)win * NHD + head) * NTOK + nrow) * HDSZ + hd;
            if (o < 128)       Q[addr] = (_Float16)(val * ATT_SCALE);
            else if (o < 256)  K[addr] = (_Float16)val;
            else               V[addr] = (_Float16)val;
        }
    }
}

// ---------------------------------------------------------------------------
// Kernel 2: flash-style window attention, one block per (win, head).
// ---------------------------------------------------------------------------
__global__ __launch_bounds__(256) void cf_k2_attn(
    const _Float16* __restrict__ Q, const _Float16* __restrict__ K,
    const _Float16* __restrict__ V, _Float16* __restrict__ O)
{
    __shared__ _Float16 Qh[NTOK * HDSZ];   // 8 KiB each
    __shared__ _Float16 Kh[NTOK * HDSZ];
    __shared__ _Float16 Vh[NTOK * HDSZ];
    __shared__ float    sc[8][16][16];     // per-wave P transpose scratch

    const int tid = threadIdx.x, bx = blockIdx.x;
    const int win = bx >> 3, head = bx & 7;
    const size_t base = (size_t)bx * NTOK * HDSZ;

    {   // async DMA fills: 3 x 512 B128 transfers, no VGPR staging
        const uint4* qs = (const uint4*)(Q + base); uint4* qd = (uint4*)Qh;
        const uint4* ks = (const uint4*)(K + base); uint4* kd = (uint4*)Kh;
        const uint4* vs = (const uint4*)(V + base); uint4* vd = (uint4*)Vh;
        for (int i = tid; i < NTOK * HDSZ / 8; i += 256) {
            async_copy_b128(qd + i, qs + i);
            async_copy_b128(kd + i, ks + i);
            async_copy_b128(vd + i, vs + i);
        }
        wait_async();
    }
    __syncthreads();

    const int wave = tid >> 5, lane = tid & 31, lr = lane & 15, hi = lane >> 4;
    for (int qi = 0; qi < 2; ++qi) {
        const int qt = wave * 2 + qi;
        v16h aq;
        for (int e = 0; e < 16; ++e) {
            const int k = kmap(e, hi);
            aq[e] = (k < HDSZ) ? Qh[(qt * 16 + lr) * HDSZ + k] : (_Float16)0.f;
        }
        float rmax[8], rsum[8];
        v8f oacc = {};
        for (int r = 0; r < 8; ++r) { rmax[r] = -1e30f; rsum[r] = 0.f; }

        for (int kt = 0; kt < 16; ++kt) {
            v16h bk;
            for (int e = 0; e < 16; ++e) {
                const int k = kmap(e, hi);
                bk[e] = (k < HDSZ) ? Kh[(kt * 16 + lr) * HDSZ + k] : (_Float16)0.f;
            }
            v8f zc = {};
            v8f s = __builtin_amdgcn_wmma_f32_16x16x32_f16(
                false, aq, false, bk, (short)0, zc, false, false);

            // online softmax: row m = r + 8*hi lives in a 16-lane group
            for (int r = 0; r < 8; ++r) {
                float sv = s[r];
                float mx = sv;
                mx = fmaxf(mx, __shfl_xor(mx, 1, 32));
                mx = fmaxf(mx, __shfl_xor(mx, 2, 32));
                mx = fmaxf(mx, __shfl_xor(mx, 4, 32));
                mx = fmaxf(mx, __shfl_xor(mx, 8, 32));
                const float mnew = fmaxf(rmax[r], mx);
                const float p = __expf(sv - mnew);
                float ps = p;
                ps += __shfl_xor(ps, 1, 32);
                ps += __shfl_xor(ps, 2, 32);
                ps += __shfl_xor(ps, 4, 32);
                ps += __shfl_xor(ps, 8, 32);
                const float corr = __expf(rmax[r] - mnew);
                rsum[r] = rsum[r] * corr + ps;
                oacc[r] *= corr;
                rmax[r] = mnew;
                sc[wave][r + 8 * hi][lr] = p;   // D-layout -> scratch[row][col]
            }
            wave_lds_sync();

            v16h ap, bv;
            for (int e = 0; e < 16; ++e) {
                const int k = kmap(e, hi);
                ap[e] = (k < 16) ? (_Float16)sc[wave][lr][k] : (_Float16)0.f;
                bv[e] = (k < 16) ? Vh[(kt * 16 + k) * HDSZ + lr] : (_Float16)0.f;
            }
            oacc = __builtin_amdgcn_wmma_f32_16x16x32_f16(
                false, ap, false, bv, (short)0, oacc, false, false);
            wave_lds_sync();
        }

        for (int r = 0; r < 8; ++r) {
            const int n = qt * 16 + r + 8 * hi;
            const float ov = oacc[r] / rsum[r];
            O[((size_t)win * NTOK + n) * CC + head * HDSZ + lr] = (_Float16)ov;
        }
    }
}

// ---------------------------------------------------------------------------
// Kernel 3: output projection (256x128 @ 128x128) + window-merge scatter.
// ---------------------------------------------------------------------------
__global__ __launch_bounds__(256) void cf_k3_proj_merge(
    const _Float16* __restrict__ O, const float* __restrict__ pw,
    const float* __restrict__ pb, float* __restrict__ Y0)
{
    __shared__ _Float16 sh[NTOK * CC];   // 64 KiB
    const int tid = threadIdx.x, win = blockIdx.x;
    {
        const uint4* src = (const uint4*)(O + (size_t)win * NTOK * CC);
        uint4* dst = (uint4*)sh;
        for (int i = tid; i < NTOK * CC / 8; i += 256)
            async_copy_b128(dst + i, src + i);
        wait_async();
    }
    __syncthreads();

    const int wave = tid >> 5, lane = tid & 31, lr = lane & 15, hi = lane >> 4;
    const int wd = win >> 6, wh = (win >> 3) & 7, ww = win & 7;
    for (int idx = wave; idx < 16 * 8; idx += 8) {
        const int rt = idx >> 3, ct = idx & 7;
        v8f acc = {};
        for (int kk = 0; kk < 4; ++kk) {
            v16h a, bf;
            for (int e = 0; e < 16; ++e) {
                const int k = kmap(e, hi);
                a[e]  = sh[(rt * 16 + lr) * CC + kk * 32 + k];
                bf[e] = (_Float16)pw[(size_t)(ct * 16 + lr) * CC + kk * 32 + k];
            }
            acc = __builtin_amdgcn_wmma_f32_16x16x32_f16(
                false, a, false, bf, (short)0, acc, false, false);
        }
        for (int r = 0; r < 8; ++r) {
            const int n = rt * 16 + r + 8 * hi;
            const int o = ct * 16 + lr;
            const int t = n >> 6, rem = n & 63;
            const int gd = rem >> 4, gh = (rem >> 2) & 3, gw = rem & 3;
            const int d = wd * GG + gd, h = wh * GG + gh, w = ww * GG + gw;
            const size_t off = ((((size_t)d * 32 + h) * 32 + w) * TT + t) * CC + o;
            Y0[off] = acc[r] + pb[o];
        }
    }
}

// ---------------------------------------------------------------------------
// Kernel 4: depthwise 3x3x3 conv over (D,H,W), T folded into batch.
// ---------------------------------------------------------------------------
__global__ __launch_bounds__(256) void cf_k4_dwconv(
    const float* __restrict__ Y0, const float* __restrict__ cw,
    const float* __restrict__ cb, float* __restrict__ Y1)
{
    const size_t idx = (size_t)blockIdx.x * 256 + threadIdx.x;  // NROW*CC
    const int c = (int)(idx & 127);
    const size_t tok = idx >> 7;           // ((d*32+h)*32+w)*4+t
    const int t = (int)(tok & 3);
    const int w = (int)((tok >> 2) & 31);
    const int h = (int)((tok >> 7) & 31);
    const int d = (int)(tok >> 12);

    float acc = cb[c];
    for (int kd = -1; kd <= 1; ++kd) {
        const int dd = d + kd; if ((unsigned)dd >= 16u) continue;
        for (int kh = -1; kh <= 1; ++kh) {
            const int hh = h + kh; if ((unsigned)hh >= 32u) continue;
            for (int kw = -1; kw <= 1; ++kw) {
                const int wp = w + kw; if ((unsigned)wp >= 32u) continue;
                const size_t src = ((((size_t)dd * 32 + hh) * 32 + wp) * TT + t) * CC + c;
                acc += Y0[src] * cw[c * 27 + (kd + 1) * 9 + (kh + 1) * 3 + (kw + 1)];
            }
        }
    }
    Y1[idx] = acc;
}

// ---------------------------------------------------------------------------
// Kernel 5: exLN -> Y2 (f32, residual base) and norm2(Y2) -> H2 (f16 MLP in).
// ---------------------------------------------------------------------------
__global__ __launch_bounds__(256) void cf_k5_lns(
    const float* __restrict__ Y1, const float* __restrict__ eg,
    const float* __restrict__ eb, const float* __restrict__ n2g,
    const float* __restrict__ n2b, float* __restrict__ Y2,
    _Float16* __restrict__ H2)
{
    const size_t tok = (size_t)blockIdx.x * 256 + threadIdx.x;
    const float* xr = Y1 + tok * CC;
    float m = 0.f, s2 = 0.f;
    for (int c = 0; c < CC; ++c) { float v = xr[c]; m += v; s2 += v * v; }
    m *= (1.f / CC);
    const float inv = rsqrtf(s2 * (1.f / CC) - m * m + 1e-5f);

    float m2 = 0.f, q2 = 0.f;
    float* yr = Y2 + tok * CC;
    for (int c = 0; c < CC; ++c) {
        const float y = (xr[c] - m) * inv * eg[c] + eb[c];
        yr[c] = y; m2 += y; q2 += y * y;
    }
    m2 *= (1.f / CC);
    const float inv2 = rsqrtf(q2 * (1.f / CC) - m2 * m2 + 1e-5f);
    _Float16* hr = H2 + tok * CC;
    for (int c = 0; c < CC; ++c)
        hr[c] = (_Float16)((yr[c] - m2) * inv2 * n2g[c] + n2b[c]);
}

// ---------------------------------------------------------------------------
// Kernel 6: fc1 GEMM (rows x 128 @ 128x512) + exact GELU -> Hm (f16).
// ---------------------------------------------------------------------------
__global__ __launch_bounds__(256) void cf_k6_fc1_gelu(
    const _Float16* __restrict__ H2, const float* __restrict__ w1,
    const float* __restrict__ b1, _Float16* __restrict__ Hm)
{
    __shared__ _Float16 sh[128 * CC];   // 32 KiB
    const int tid = threadIdx.x;
    const size_t rowBase = (size_t)blockIdx.x * 128;
    {
        const uint4* src = (const uint4*)(H2 + rowBase * CC);
        uint4* dst = (uint4*)sh;
        for (int i = tid; i < 128 * CC / 8; i += 256)
            async_copy_b128(dst + i, src + i);
        wait_async();
    }
    __syncthreads();

    const int wave = tid >> 5, lane = tid & 31, lr = lane & 15, hi = lane >> 4;
    for (int idx = wave; idx < 8 * 32; idx += 8) {
        const int rt = idx >> 5, ct = idx & 31;
        v8f acc = {};
        for (int kk = 0; kk < 4; ++kk) {
            v16h a, bf;
            for (int e = 0; e < 16; ++e) {
                const int k = kmap(e, hi);
                a[e]  = sh[(rt * 16 + lr) * CC + kk * 32 + k];
                bf[e] = (_Float16)w1[(size_t)(ct * 16 + lr) * CC + kk * 32 + k];
            }
            acc = __builtin_amdgcn_wmma_f32_16x16x32_f16(
                false, a, false, bf, (short)0, acc, false, false);
        }
        for (int r = 0; r < 8; ++r) {
            const size_t n = rowBase + rt * 16 + r + 8 * hi;
            const int o = ct * 16 + lr;
            const float v = acc[r] + b1[o];
            const float ge = 0.5f * v * (1.f + erff(v * 0.70710678f));
            Hm[n * MLPH + o] = (_Float16)ge;
        }
    }
}

// ---------------------------------------------------------------------------
// Kernel 7: fc2 GEMM (rows x 512 @ 512x128) + bias + residual -> d_out (f32).
// ---------------------------------------------------------------------------
__global__ __launch_bounds__(256) void cf_k7_fc2_res(
    const _Float16* __restrict__ Hm, const float* __restrict__ w2,
    const float* __restrict__ b2, const float* __restrict__ Y2,
    float* __restrict__ out)
{
    const int tid = threadIdx.x;
    const size_t rowBase = (size_t)blockIdx.x * 128;
    const int wave = tid >> 5, lane = tid & 31, lr = lane & 15, hi = lane >> 4;
    for (int idx = wave; idx < 8 * 8; idx += 8) {
        const int rt = idx >> 3, ct = idx & 7;
        v8f acc = {};
        for (int kk = 0; kk < 16; ++kk) {
            // hint the streamed weight row for the next K step into L2
            if (kk + 1 < 16)
                __builtin_prefetch(&w2[(size_t)(ct * 16 + lr) * MLPH + (kk + 1) * 32], 0, 1);
            v16h a, bf;
            for (int e = 0; e < 16; ++e) {
                const int k = kmap(e, hi);
                a[e]  = Hm[(rowBase + rt * 16 + lr) * MLPH + kk * 32 + k];
                bf[e] = (_Float16)w2[(size_t)(ct * 16 + lr) * MLPH + kk * 32 + k];
            }
            acc = __builtin_amdgcn_wmma_f32_16x16x32_f16(
                false, a, false, bf, (short)0, acc, false, false);
        }
        for (int r = 0; r < 8; ++r) {
            const size_t n = rowBase + rt * 16 + r + 8 * hi;
            const int o = ct * 16 + lr;
            out[n * CC + o] = acc[r] + b2[o] + Y2[n * CC + o];
        }
    }
}

// ---------------------------------------------------------------------------
// Host-side orchestration. Workspace plan (peak 128 MiB, no live overlaps):
//   [0,16M)   Q f16      -> later Y0 f32 [0,32M) -> later Y2 f32 [0,32M)
//   [16,32M)  K f16
//   [32,48M)  V f16      -> later H2 f16 [32,48M)
//   [48,64M)  O f16
//   [64,96M)  Y1 f32     -> later Hm f16 [64,128M)
// ---------------------------------------------------------------------------
extern "C" void kernel_launch(void* const* d_in, const int* in_sizes, int n_in,
                              void* d_out, int out_size, void* d_ws, size_t ws_size,
                              hipStream_t stream) {
    (void)in_sizes; (void)n_in; (void)out_size; (void)ws_size;
    const float* x       = (const float*)d_in[0];
    const float* norm1_g = (const float*)d_in[1];
    const float* norm1_b = (const float*)d_in[2];
    const float* pos     = (const float*)d_in[3];
    const float* temb    = (const float*)d_in[4];
    const float* qkv_w   = (const float*)d_in[5];
    const float* qkv_b   = (const float*)d_in[6];
    const float* proj_w  = (const float*)d_in[7];
    const float* proj_b  = (const float*)d_in[8];
    const float* conv_w  = (const float*)d_in[9];
    const float* conv_b  = (const float*)d_in[10];
    const float* exln_g  = (const float*)d_in[11];
    const float* exln_b  = (const float*)d_in[12];
    const float* norm2_g = (const float*)d_in[13];
    const float* norm2_b = (const float*)d_in[14];
    const float* fc1_w   = (const float*)d_in[15];
    const float* fc1_b   = (const float*)d_in[16];
    const float* fc2_w   = (const float*)d_in[17];
    const float* fc2_b   = (const float*)d_in[18];

    char* ws = (char*)d_ws;
    const size_t MB = 1u << 20;
    _Float16* Q  = (_Float16*)(ws + 0 * MB);
    _Float16* K  = (_Float16*)(ws + 16 * MB);
    _Float16* V  = (_Float16*)(ws + 32 * MB);
    _Float16* O  = (_Float16*)(ws + 48 * MB);
    float*    Y0 = (float*)   (ws + 0 * MB);
    float*    Y1 = (float*)   (ws + 64 * MB);
    float*    Y2 = (float*)   (ws + 0 * MB);
    _Float16* H2 = (_Float16*)(ws + 32 * MB);
    _Float16* Hm = (_Float16*)(ws + 64 * MB);
    float*    out = (float*)d_out;

    cf_k1_ln_pe_qkv<<<NWIN, 256, 0, stream>>>(x, norm1_g, norm1_b, pos, temb,
                                              qkv_w, qkv_b, Q, K, V);
    cf_k2_attn<<<NWIN * NHD, 256, 0, stream>>>(Q, K, V, O);
    cf_k3_proj_merge<<<NWIN, 256, 0, stream>>>(O, proj_w, proj_b, Y0);
    cf_k4_dwconv<<<(NROW * CC) / 256, 256, 0, stream>>>(Y0, conv_w, conv_b, Y1);
    cf_k5_lns<<<NROW / 256, 256, 0, stream>>>(Y1, exln_g, exln_b,
                                              norm2_g, norm2_b, Y2, H2);
    cf_k6_fc1_gelu<<<NROW / 128, 256, 0, stream>>>(H2, fc1_w, fc1_b, Hm);
    cf_k7_fc2_res<<<NROW / 128, 256, 0, stream>>>(Hm, fc2_w, fc2_b, Y2, out);
}
